// NSE_64226940944435
// MI455X (gfx1250) — matrix operand order repaired and checked by reading, not compiled, not run on hardware
//
#include <hip/hip_runtime.h>
#include <hip/hip_bf16.h>

typedef __attribute__((ext_vector_type(16))) __bf16 v16bf;
typedef __attribute__((ext_vector_type(8)))  __bf16 v8bf;
typedef __attribute__((ext_vector_type(8)))  float  v8f;

#define T_  256
#define D_  256
#define NWG 16     // workgroups; each owns BB batch rows (16*2 = 32 = B)
#define BB  2      // batches per workgroup (valid rows inside the 16-row M tile)
#define NT  1024   // threads per workgroup = 32 wave32s (8 waves / SIMD32)

__device__ __forceinline__ float hsig(float x) {
    return fminf(fmaxf(0.2f * x + 0.5f, 0.0f), 1.0f);
}

// One 16x16 output tile (single M tile row) of C = A x Bt + bias.
// CDNA5 WMMA 16x16x32 bf16 operand layouts (ISA 7.12.2):
//   A: lane L -> row L%16; ks = 8*(L/16)+{0..7} and +16..+23  (two 16B LDS runs)
//   B: lane L -> col L%16; ks = 8*(L/16)+{0..7} and +16..+23  (two 16B global runs)
//   C: lane L -> col L%16; VGPR v -> row v + 8*(L/16); store only rows < `rows`
__device__ __forceinline__ void gemm_tile(
    const __bf16* A, int As,
    const __bf16* __restrict__ Bt, int Bs,
    const float* __restrict__ bias,
    float* C, int Cs, int rows,
    int tn, int K, int lane)
{
    const int r = lane & 15;
    const int g = lane >> 4;
    v8f acc = {};
    const __bf16* ap = A + r * As + 8 * g;
    const __bf16* bp = Bt + (tn * 16 + r) * Bs + 8 * g;
    #pragma unroll 4
    for (int k0 = 0; k0 < K; k0 += 32) {
        v16bf va, vb;
        ((v8bf*)&va)[0] = *(const v8bf*)(ap + k0);
        ((v8bf*)&va)[1] = *(const v8bf*)(ap + k0 + 16);
        ((v8bf*)&vb)[0] = *(const v8bf*)(bp + k0);
        ((v8bf*)&vb)[1] = *(const v8bf*)(bp + k0 + 16);
        acc = __builtin_amdgcn_wmma_f32_16x16x32_bf16(
            false, va, false, vb, (short)0, acc, false, false);
    }
    const float bv = bias[tn * 16 + r];
    float* cp = C + tn * 16 + r;
    #pragma unroll
    for (int v = 0; v < 8; ++v) {
        int m = v + 8 * g;
        if (m < rows) cp[m * Cs] = acc[v] + bv;   // predicate: valid batch rows only
    }
}

// One-time prep: mem = x (f32); weights -> bf16, transposed to [N][K], with
// [W;U] concatenated along K so each LSTM GEMM is a single K-loop.
__global__ void nse_convert(const float* __restrict__ x,
                            const float* __restrict__ rW, const float* __restrict__ rU,
                            const float* __restrict__ wW, const float* __restrict__ wU,
                            const float* __restrict__ cW,
                            float* __restrict__ mem,
                            __bf16* __restrict__ rWU, __bf16* __restrict__ wWU,
                            __bf16* __restrict__ cWt)
{
    const int stride = gridDim.x * blockDim.x;
    const int i0 = blockIdx.x * blockDim.x + threadIdx.x;
    for (int i = i0; i < 2097152; i += stride) mem[i] = x[i];
    for (int i = i0; i < 524288; i += stride) {          // reader [1024][512]
        int n = i >> 9, k = i & 511;
        float v = (k < 256) ? rW[k * 1024 + n] : rU[(k - 256) * 1024 + n];
        rWU[i] = (__bf16)v;
    }
    for (int i = i0; i < 786432; i += stride) {          // writer [1024][768]
        int n = i / 768, k = i - n * 768;
        float v = (k < 512) ? wW[k * 1024 + n] : wU[(k - 512) * 1024 + n];
        wWU[i] = (__bf16)v;
    }
    for (int i = i0; i < 262144; i += stride) {          // composer [512][512]
        int n = i >> 9, k = i & 511;
        cWt[i] = (__bf16)cW[k * 512 + n];
    }
}

// 16 independent persistent workgroups; WG `wg` owns batches wg*BB..wg*BB+BB-1
// and runs the entire 512-step recurrence privately (workgroup barriers only).
__global__ void __launch_bounds__(NT)
nse_persistent(const float* __restrict__ x,
               const float* __restrict__ rb, const float* __restrict__ wb,
               const float* __restrict__ cb,
               const __bf16* __restrict__ rWU, const __bf16* __restrict__ wWU,
               const __bf16* __restrict__ cWt,
               float* __restrict__ mem, float* __restrict__ oseq,
               float* __restrict__ out)
{
    __shared__ float zbuf[BB * 1024];                  //  8 KB  gate pre-activations
    __shared__ float hbuf[BB * 256];                   //  2 KB
    __shared__ float cbuf[BB * 256];                   //  2 KB
    __shared__ __align__(16) __bf16 abf[16 * 768];     // 24 KB  A tile (rows >= BB garbage, never stored)
    __shared__ float sbuf[BB * 256];                   //  2 KB  scores / softmax z
    __shared__ float pbuf[NT];                         //  4 KB  partial-dot scratch

    const int tid  = threadIdx.x;
    const int lane = tid & 31;
    const int wv   = tid >> 5;
    const int bg0  = blockIdx.x * BB;

    if (tid < BB * 256) { hbuf[tid] = 0.f; cbuf[tid] = 0.f; }
    __syncthreads();

    // ---------------- reader LSTM scan (this WG's batches) ----------------
    for (int t = 0; t < T_; ++t) {
        if (tid < BB * 256) {                 // stage A = [x_t | h] as bf16
            int lb = tid >> 8, d = tid & 255;
            abf[lb * 768 + d]       = (__bf16)x[((bg0 + lb) * T_ + t) * D_ + d];
            abf[lb * 768 + 256 + d] = (__bf16)hbuf[tid];
        }
        __syncthreads();
        #pragma unroll
        for (int j = 0; j < 2; ++j)           // z = [x|h] @ [W;U] + b  (K=512, 64 N-tiles)
            gemm_tile(abf, 768, rWU, 512, rb, zbuf, 1024, BB, wv + 32 * j, 512, lane);
        __syncthreads();
        if (tid < BB * 256) {                 // gates, state update
            int lb = tid >> 8, d = tid & 255;
            const float* zr = zbuf + lb * 1024 + d;
            float ii = hsig(zr[0]), ff = hsig(zr[256]);
            float gg = tanhf(zr[512]), oo = hsig(zr[768]);
            float c = ff * cbuf[tid] + ii * gg;
            float h = oo * tanhf(c);
            cbuf[tid] = c; hbuf[tid] = h;
            oseq[(bg0 + lb) * 65536 + t * 256 + d] = h;
        }
        __threadfence_block();
        __syncthreads();
    }

    if (tid < BB * 256) { hbuf[tid] = 0.f; cbuf[tid] = 0.f; }
    __syncthreads();

    // ---------------- writer loop ----------------
    for (int s = 0; s < T_; ++s) {
        if (tid < BB * 256) {                 // stage o_t: f32 in zbuf, bf16 in abf[0..255]
            int lb = tid >> 8, d = tid & 255;
            float ov = oseq[(bg0 + lb) * 65536 + s * 256 + d];
            zbuf[tid] = ov;
            abf[lb * 768 + d] = (__bf16)ov;
        }
        __syncthreads();
        {   // scores[lb][t] = o . mem[lb][t], 2-way d-split so all 1024 lanes stream
            int q = tid >> 1, half = tid & 1;
            int lb = q >> 8, t = q & 255;
            const float* ob = zbuf + lb * 256 + half * 128;
            const float* mr = mem + (bg0 + lb) * 65536 + t * 256 + half * 128;
            float acc = 0.f;
            #pragma unroll 4
            for (int d = 0; d < 128; ++d) acc = fmaf(ob[d], mr[d], acc);
            pbuf[tid] = acc;
        }
        __syncthreads();
        if (tid < BB * 256) sbuf[tid] = pbuf[2 * tid] + pbuf[2 * tid + 1];
        __syncthreads();
        if (wv < BB) {                        // softmax: one wave32 per batch row
            float v[8]; float mx = -3.4e38f;
            #pragma unroll
            for (int i = 0; i < 8; ++i) { v[i] = sbuf[wv * 256 + lane + 32 * i]; mx = fmaxf(mx, v[i]); }
            for (int off = 16; off > 0; off >>= 1) mx = fmaxf(mx, __shfl_xor(mx, off, 32));
            float sum = 0.f;
            #pragma unroll
            for (int i = 0; i < 8; ++i) { v[i] = expf(v[i] - mx); sum += v[i]; }
            for (int off = 16; off > 0; off >>= 1) sum += __shfl_xor(sum, off, 32);
            float inv = 1.0f / sum;
            #pragma unroll
            for (int i = 0; i < 8; ++i) sbuf[wv * 256 + lane + 32 * i] = v[i] * inv;
        }
        __syncthreads();
        {   // m_rt[lb][d] = sum_t z[lb][t]*mem[lb][t][d], 2-way t-split
            int q = tid >> 1, half = tid & 1;
            int lb = q >> 8, d = q & 255;
            const float* zr = sbuf + lb * 256 + half * 128;
            const float* mc = mem + (bg0 + lb) * 65536 + half * 128 * 256 + d;
            float acc = 0.f;
            #pragma unroll 4
            for (int t = 0; t < 128; ++t) acc = fmaf(zr[t], mc[t * 256], acc);
            pbuf[tid] = acc;
        }
        __syncthreads();
        if (tid < BB * 256) {
            int lb = tid >> 8, d = tid & 255;
            abf[lb * 768 + 256 + d] = (__bf16)(pbuf[2 * tid] + pbuf[2 * tid + 1]);
        }
        __syncthreads();
        // composer: ct = [o|m] @ cW + cb  (K=512, 32 N-tiles -> 1 per wave)
        gemm_tile(abf, 768, cWt, 512, cb, zbuf, 1024, BB, wv, 512, lane);
        __syncthreads();
        if (tid < BB * 512) {                 // abf[0..511] = bf16(ct)
            int lb = tid >> 9, k = tid & 511;
            abf[lb * 768 + k] = (__bf16)zbuf[lb * 1024 + k];
        }
        if (tid < BB * 256) {                 // abf[512..767] = bf16(wh)
            int lb = tid >> 8, d = tid & 255;
            abf[lb * 768 + 512 + d] = (__bf16)hbuf[tid];
        }
        __syncthreads();
        #pragma unroll
        for (int j = 0; j < 2; ++j)           // z = [ct|wh] @ [wW;wU] + wb (K=768)
            gemm_tile(abf, 768, wWU, 768, wb, zbuf, 1024, BB, wv + 32 * j, 768, lane);
        __syncthreads();
        if (tid < BB * 256) {                 // writer LSTM gates
            int lb = tid >> 8, d = tid & 255;
            const float* zr = zbuf + lb * 1024 + d;
            float ii = hsig(zr[0]), ff = hsig(zr[256]);
            float gg = tanhf(zr[512]), oo = hsig(zr[768]);
            float c = ff * cbuf[tid] + ii * gg;
            float h = oo * tanhf(c);
            cbuf[tid] = c; hbuf[tid] = h;
        }
        __syncthreads();
        {   // mem = mem + z*(wh - mem) over this WG's slice (coalesced)
            float* mw = mem + bg0 * 65536;
            #pragma unroll 4
            for (int i = 0; i < (BB * 65536) / NT; ++i) {
                int e = tid + NT * i;
                int lb = e >> 16, t = (e >> 8) & 255, d = e & 255;
                float z = sbuf[lb * 256 + t];
                float m = mw[e];
                mw[e] = m + z * (hbuf[lb * 256 + d] - m);
            }
        }
        __threadfence_block();
        __syncthreads();
    }

    if (tid < BB * 256) out[bg0 * 256 + tid] = hbuf[tid];
}

extern "C" void kernel_launch(void* const* d_in, const int* in_sizes, int n_in,
                              void* d_out, int out_size, void* d_ws, size_t ws_size,
                              hipStream_t stream) {
    const float* x  = (const float*)d_in[0];
    const float* rW = (const float*)d_in[1];
    const float* rU = (const float*)d_in[2];
    const float* rb = (const float*)d_in[3];
    const float* wW = (const float*)d_in[4];
    const float* wU = (const float*)d_in[5];
    const float* wb = (const float*)d_in[6];
    const float* cW = (const float*)d_in[7];
    const float* cb = (const float*)d_in[8];

    char* ws = (char*)d_ws;
    float*  mem  = (float*)(ws);                 // 8 MB   f32 [32][256][256]
    float*  oseq = (float*)(ws + 8388608);       // 8 MB   f32 [32][256][256]
    __bf16* rWU  = (__bf16*)(ws + 16777216);     // 1 MB   bf16 [1024][512]
    __bf16* wWU  = (__bf16*)(ws + 17825792);     // 1.5 MB bf16 [1024][768]
    __bf16* cWt  = (__bf16*)(ws + 19398656);     // 0.5 MB bf16 [512][512]

    nse_convert<<<2048, 256, 0, stream>>>(x, rW, rU, wW, wU, cW, mem, rWU, wWU, cWt);
    nse_persistent<<<NWG, NT, 0, stream>>>(x, rb, wb, cb, rWU, wWU, cWt,
                                           mem, oseq, (float*)d_out);
}
